// Qwen3OmniMoeThinkerTextSparseMoeBlock_38920993636818
// MI455X (gfx1250) — matrix operand order repaired and checked by reading, not compiled, not run on hardware
//
#include <hip/hip_runtime.h>

// ---------------------------------------------------------------------------
// Qwen3-Omni MoE block, sparse top-8 formulation, bf16 WMMA on gfx1250.
// Dims (hardcoded to the reference): B=2,S=2048 -> T=4096, H=2048, E=32, I=768
// Roofline: 604 MB fp32 expert weights stream from HBM once -> ~26us floor at
// 23.3 TB/s; 309 GFLOP of sparse matmul is far below any WMMA ceiling, so the
// kernel is bandwidth-bound and we optimize data movement: async LDS staging,
// L2-resident weight streaming, one-pass per-expert token grouping.
// ---------------------------------------------------------------------------

#define T_TOK 4096
#define H_DIM 2048
#define E_EXP 32
#define I_DIM 768
#define TOPK  8

typedef __attribute__((ext_vector_type(16))) __bf16 v16bf;
typedef __attribute__((ext_vector_type(8)))  __bf16 v8bf;
typedef __attribute__((ext_vector_type(8)))  float  v8f;

// Exact pointee type expected by the gfx1250 async-LDS builtins (from the
// compiler diagnostic): 16-byte int vector.
typedef int async_b128_t __attribute__((vector_size(16)));

// LDS leading dims padded so row stride in dwords % 64 == 4 -> the 16-lane
// row gathers hit distinct bank groups (conflict-free).
#define XF_CH 256   // K-chunk of X staged per buffer (f32)
#define XF_LD 260   // 256 + 4 f32 pad
#define H2_LD 776   // 768 + 8 bf16 pad

#if __has_builtin(__builtin_amdgcn_global_load_async_to_lds_b128)
#define HAVE_ASYNC_LDS 1
#else
#define HAVE_ASYNC_LDS 0
#endif

#define AS1P(p) ((__attribute__((address_space(1))) async_b128_t*)(p))
#define AS3P(p) ((__attribute__((address_space(3))) async_b128_t*)(p))

// ---------------------------------------------------------------------------
// Kernel 1: router. One block per token; wave0 does softmax+top8 with E=32
// experts mapped 1:1 onto the 32 lanes of a CDNA5 wave.
// ---------------------------------------------------------------------------
__global__ void __launch_bounds__(256) moe_router_kernel(
    const float* __restrict__ x, const float* __restrict__ gate_w,
    int* __restrict__ topk_idx, float* __restrict__ topk_w)
{
  const int t    = blockIdx.x;
  const int lane = threadIdx.x & 31;
  const int wave = threadIdx.x >> 5;
  __shared__ float logits[E_EXP];

  const float* xr = x + (size_t)t * H_DIM;
  for (int ee = 0; ee < 4; ++ee) {
    const int e = wave * 4 + ee;
    const float* gr = gate_w + (size_t)e * H_DIM;
    float acc = 0.f;
    for (int h = lane; h < H_DIM; h += 32) acc += xr[h] * gr[h];
#pragma unroll
    for (int off = 16; off >= 1; off >>= 1) acc += __shfl_xor(acc, off, 32);
    if (lane == 0) logits[e] = acc;
  }
  __syncthreads();

  if (wave == 0) {
    float v = logits[lane];           // lane == expert index
    float m = v;
#pragma unroll
    for (int off = 16; off >= 1; off >>= 1) m = fmaxf(m, __shfl_xor(m, off, 32));
    float ex = __expf(v - m);
    float s = ex;
#pragma unroll
    for (int off = 16; off >= 1; off >>= 1) s += __shfl_xor(s, off, 32);
    float p  = ex / s;
    float pw = p;

    float topv[TOPK];
    int   topi[TOPK];
    float selsum = 0.f;
#pragma unroll
    for (int k = 0; k < TOPK; ++k) {
      float mv = pw; int mi = lane;
#pragma unroll
      for (int off = 16; off >= 1; off >>= 1) {
        float ov = __shfl_xor(mv, off, 32);
        int   oi = __shfl_xor(mi, off, 32);
        if (ov > mv || (ov == mv && oi < mi)) { mv = ov; mi = oi; }
      }
      topv[k] = mv; topi[k] = mi; selsum += mv;
      if (lane == mi) pw = -1.0f;
    }
    const float denom = fmaxf(selsum, 1e-9f);
    if (lane == 0) {
#pragma unroll
      for (int k = 0; k < TOPK; ++k) {
        topk_idx[t * TOPK + k] = topi[k];
        topk_w  [t * TOPK + k] = topv[k] / denom;
      }
    }
  }
}

// ---------------------------------------------------------------------------
// Kernel 2: build per-expert assignment lists. Assignment id a = t*8 + k.
// ---------------------------------------------------------------------------
__global__ void moe_scatter_kernel(const int* __restrict__ topk_idx,
                                   int* __restrict__ cnt, int* __restrict__ list)
{
  const int a = blockIdx.x * blockDim.x + threadIdx.x;
  if (a >= T_TOK * TOPK) return;
  const int e   = topk_idx[a];
  const int pos = atomicAdd(&cnt[e], 1);
  list[e * T_TOK + pos] = a;
}

// ---------------------------------------------------------------------------
// Fragment helpers (bf16 16x16x32).
// A (16x32, 16-bit): lanes 0-15 row M=lane, K k0..k0+15;
// lanes 16-31 same rows, K k0+16..k0+31.
// ---------------------------------------------------------------------------
__device__ __forceinline__ v16bf a_frag_from_f32(const float* rowbase, int k0, int half)
{
  const float4 a0 = *(const float4*)(rowbase + k0 + 16 * half);
  const float4 a1 = *(const float4*)(rowbase + k0 + 16 * half + 4);
  const float4 a2 = *(const float4*)(rowbase + k0 + 16 * half + 8);
  const float4 a3 = *(const float4*)(rowbase + k0 + 16 * half + 12);
  union { v16bf v; __bf16 e[16]; } A;
  A.e[0]  = (__bf16)a0.x; A.e[1]  = (__bf16)a0.y; A.e[2]  = (__bf16)a0.z; A.e[3]  = (__bf16)a0.w;
  A.e[4]  = (__bf16)a1.x; A.e[5]  = (__bf16)a1.y; A.e[6]  = (__bf16)a1.z; A.e[7]  = (__bf16)a1.w;
  A.e[8]  = (__bf16)a2.x; A.e[9]  = (__bf16)a2.y; A.e[10] = (__bf16)a2.z; A.e[11] = (__bf16)a2.w;
  A.e[12] = (__bf16)a3.x; A.e[13] = (__bf16)a3.y; A.e[14] = (__bf16)a3.z; A.e[15] = (__bf16)a3.w;
  return A.v;
}

__device__ __forceinline__ v16bf a_frag_from_bf16(const __bf16* rowbase, int k0, int half)
{
  union { v16bf v; v8bf h[2]; } A;
  A.h[0] = *(const v8bf*)(rowbase + k0 + 16 * half);
  A.h[1] = *(const v8bf*)(rowbase + k0 + 16 * half + 8);
  return A.v;
}

// B (32x16, 16-bit): lane n<16 -> column n, K = kg..kg+15; lanes 16-31 same
// columns, K = kg+16..kg+31. Each strided dword load is 64B-contiguous across
// a half-wave -> coalesced; f32 -> bf16 packs via v_cvt_pk_bf16_f32.
__device__ __forceinline__ v16bf load_b_frag(const float* __restrict__ w,
                                             int kg, int ld, int col, int khalf)
{
  union { v16bf v; __bf16 e[16]; } B;
  const float* p = w + (size_t)(kg + khalf) * ld + col;
#pragma unroll
  for (int j = 0; j < 16; ++j) B.e[j] = (__bf16)p[(size_t)j * ld];
  return B.v;
}

// Stage one 16 x 256-f32 X chunk into LDS. Async (gfx1250 ASYNCcnt path) when
// the builtin exists, plain vector copy otherwise. 1024 16B granules, 4/thread.
__device__ __forceinline__ void stage_x_chunk(float (*buf)[XF_LD],
                                              const float* __restrict__ x,
                                              const int* tokS, int kbase, int tid)
{
#pragma unroll
  for (int j = 0; j < 4; ++j) {
    const int cidx = tid + 256 * j;       // coalesced granule index
    const int row  = cidx >> 6;           // 64 granules per row
    const int col  = (cidx & 63) * 4;
    const float* gp = x + (size_t)tokS[row] * H_DIM + kbase + col;
    float* lp = &buf[row][col];
#if HAVE_ASYNC_LDS
    __builtin_amdgcn_global_load_async_to_lds_b128(AS1P(gp), AS3P(lp), 0, 0);
#else
    *(float4*)lp = *(const float4*)gp;
#endif
  }
}

__device__ __forceinline__ void wait_async_lds()
{
  asm volatile("s_wait_asynccnt 0" ::: "memory");
}

// ---------------------------------------------------------------------------
// Kernel 3: fused expert FFN for a tile of 16 tokens of one expert.
//   h2 = silu(X Wg) * (X Wu)   [16 x 768, bf16 in LDS]
//   y  = h2 Wd                 [16 x 2048]
//   out[t] += route_w * y      (fp32 atomics)
// 256 threads = 8 waves; block-uniform early exit keeps EXEC all-1s at WMMAs.
// X is staged through a double-buffered LDS region with async copies so the
// next chunk's HBM->LDS transfer overlaps the current chunk's WMMAs.
// ---------------------------------------------------------------------------
__global__ void __launch_bounds__(256) moe_expert_kernel(
    const float* __restrict__ x,
    const float* __restrict__ w_gate, const float* __restrict__ w_up,
    const float* __restrict__ w_down,
    const float* __restrict__ topk_w,
    const int* __restrict__ list, const int* __restrict__ cnt,
    float* __restrict__ out)
{
  const int e         = blockIdx.y;
  const int nt        = cnt[e];
  const int tile_base = blockIdx.x * 16;
  if (tile_base >= nt) return;

  __shared__ __align__(16) float  xf32[2][16][XF_LD];   // 33.3 KB
  __shared__ __align__(16) __bf16 h2s[16][H2_LD];       // 24.8 KB
  __shared__ int   tokS[16];
  __shared__ float wS[16];

  const int tid  = threadIdx.x;
  const int lane = tid & 31;
  const int wave = tid >> 5;

  if (tid < 16) {
    int ridx = tile_base + tid;
    if (ridx > nt - 1) ridx = nt - 1;      // clamped rows never scattered
    const int a = list[e * T_TOK + ridx];
    tokS[tid] = a >> 3;                    // a = t*8 + k
    wS[tid]   = topk_w[a];
  }
  __syncthreads();

  const float* wg = w_gate + (size_t)e * H_DIM * I_DIM;
  const float* wu = w_up   + (size_t)e * H_DIM * I_DIM;
  const float* wd = w_down + (size_t)e * I_DIM * H_DIM;

  const int n     = lane & 15;
  const int half  = lane >> 4;
  const int khalf = half * 16;

  const v8f vzero = {};
  v8f accg[6], accu[6];
#pragma unroll
  for (int m = 0; m < 6; ++m) { accg[m] = vzero; accu[m] = vzero; }

  // ---- Phase A: gate & up GEMMs over K=2048, 8 double-buffered chunks -----
  stage_x_chunk(xf32[0], x, tokS, 0, tid);
  wait_async_lds();
  __syncthreads();

  for (int kc = 0; kc < H_DIM / XF_CH; ++kc) {
    const int cur = kc & 1;
    if (kc + 1 < H_DIM / XF_CH)
      stage_x_chunk(xf32[cur ^ 1], x, tokS, (kc + 1) * XF_CH, tid);

    const float* xrow = &xf32[cur][n][0];
    for (int ks = 0; ks < XF_CH / 32; ++ks) {
      const int k0 = ks * 32;
      const int kg = kc * XF_CH + k0;
      const v16bf afrag = a_frag_from_f32(xrow, k0, half);  // 1 frag / 12 WMMAs
      if (kg + 32 < H_DIM) {
        __builtin_prefetch(wg + (size_t)(kg + 32 + khalf) * I_DIM + n, 0, 1);
        __builtin_prefetch(wu + (size_t)(kg + 32 + khalf) * I_DIM + n, 0, 1);
      }
#pragma unroll
      for (int m = 0; m < 6; ++m) {        // 8 waves x 6 = 48 N-tiles = I/16
        const int i0 = (wave + 8 * m) * 16;
        const v16bf bg = load_b_frag(wg, kg, I_DIM, i0 + n, khalf);
        accg[m] = __builtin_amdgcn_wmma_f32_16x16x32_bf16(
            false, afrag, false, bg, (short)0, accg[m], false, false);
        const v16bf bu = load_b_frag(wu, kg, I_DIM, i0 + n, khalf);
        accu[m] = __builtin_amdgcn_wmma_f32_16x16x32_bf16(
            false, afrag, false, bu, (short)0, accu[m], false, false);
      }
    }
    wait_async_lds();     // next chunk's async copies landed during compute
    __syncthreads();
  }

  // ---- silu(g) * u -> h2s (bf16) ----
#pragma unroll
  for (int m = 0; m < 6; ++m) {
    const int i0 = (wave + 8 * m) * 16;
#pragma unroll
    for (int v = 0; v < 8; ++v) {
      const float g = accg[m][v];
      const float u = accu[m][v];
      const float s = g / (1.0f + __expf(-g));   // silu
      const int row = v + 8 * half;              // C/D: M = v + 8*half, N = lane&15
      h2s[row][i0 + n] = (__bf16)(s * u);
    }
  }
  __syncthreads();

  // ---- Phase B: down projection, K = 768, N = 2048 in two register groups --
#pragma unroll 1
  for (int g = 0; g < 2; ++g) {
    v8f acc2[8];
#pragma unroll
    for (int mm = 0; mm < 8; ++mm) acc2[mm] = vzero;
    for (int ks = 0; ks < I_DIM / 32; ++ks) {
      const int k0 = ks * 32;
      const v16bf afrag = a_frag_from_bf16(&h2s[n][0], k0, half); // 1 frag / 8 WMMAs
      if (k0 + 32 < I_DIM)
        __builtin_prefetch(wd + (size_t)(k0 + 32 + khalf) * H_DIM + n, 0, 1);
#pragma unroll
      for (int mm = 0; mm < 8; ++mm) {     // 8 waves x 16 = 128 N-tiles = H/16
        const int h0 = (wave + 8 * (g * 8 + mm)) * 16;
        const v16bf bd = load_b_frag(wd, k0, H_DIM, h0 + n, khalf);
        acc2[mm] = __builtin_amdgcn_wmma_f32_16x16x32_bf16(
            false, afrag, false, bd, (short)0, acc2[mm], false, false);
      }
    }
#pragma unroll
    for (int mm = 0; mm < 8; ++mm) {
      const int h0 = (wave + 8 * (g * 8 + mm)) * 16;
#pragma unroll
      for (int v = 0; v < 8; ++v) {
        const int row = v + 8 * half;
        if (tile_base + row < nt) {
          atomicAdd(&out[(size_t)tokS[row] * H_DIM + h0 + n], wS[row] * acc2[mm][v]);
        }
      }
    }
  }
}

// ---------------------------------------------------------------------------
// Launch. Workspace layout (bytes):
//   [0,       131072)  topk_idx  int[T*8]
//   [131072,  262144)  topk_w    float[T*8]
//   [262144,  262272)  cnt       int[E]
//   [262400,  786688)  list      int[E*T]
// ---------------------------------------------------------------------------
extern "C" void kernel_launch(void* const* d_in, const int* in_sizes, int n_in,
                              void* d_out, int out_size, void* d_ws, size_t ws_size,
                              hipStream_t stream)
{
  (void)in_sizes; (void)n_in; (void)out_size; (void)ws_size;
  const float* x      = (const float*)d_in[0];
  const float* gate_w = (const float*)d_in[1];
  const float* w_gate = (const float*)d_in[2];
  const float* w_up   = (const float*)d_in[3];
  const float* w_down = (const float*)d_in[4];
  float* out = (float*)d_out;

  char*  ws       = (char*)d_ws;
  int*   topk_idx = (int*)  (ws + 0);
  float* topk_w   = (float*)(ws + 131072);
  int*   cnt      = (int*)  (ws + 262144);
  int*   list     = (int*)  (ws + 262400);

  (void)hipMemsetAsync(out, 0, (size_t)T_TOK * H_DIM * sizeof(float), stream);
  (void)hipMemsetAsync(cnt, 0, E_EXP * sizeof(int), stream);

  moe_router_kernel<<<T_TOK, 256, 0, stream>>>(x, gate_w, topk_idx, topk_w);
  moe_scatter_kernel<<<(T_TOK * TOPK + 255) / 256, 256, 0, stream>>>(topk_idx, cnt, list);

  dim3 grid(T_TOK / 16, E_EXP);
  moe_expert_kernel<<<grid, 256, 0, stream>>>(x, w_gate, w_up, w_down,
                                              topk_w, list, cnt, out);
}
